// SpikingWANN_66494683676773
// MI455X (gfx1250) — 8-bit matrix operand order repaired and checked by reading, compiled, not run on hardware
//
#include <hip/hip_runtime.h>
#include <hip/hip_bf16.h>


#define Ldim 4
#define Hdim 2048
#define Bdim 256
#define Kfan 64
#define Tdim 25
#define Odim 10
#define KOdim 512

typedef __attribute__((ext_vector_type(8))) int v8i;
typedef unsigned char ubyte;

#define OFF_W   0u
#define OFF_A0  16777216u
#define OFF_A1  17301504u
#define OFF_SP  17825792u
#define OFF_VH  18350080u
#define OFF_VO  26738688u
#define WS_TOTAL 26755072u

__global__ void wann_zero(uint4* __restrict__ ws, float* __restrict__ out,
                          int n16, int nout) {
    int id = blockIdx.x * 256 + threadIdx.x;
    if (id < n16) { uint4 z{}; ws[id] = z; }
    if (id < nout) out[id] = 0.0f;
}

__global__ void wann_densify(const int* __restrict__ esrc,
                             const float* __restrict__ ew,
                             signed char* __restrict__ W) {
    int id = blockIdx.x * 256 + threadIdx.x;
    int l = id >> 11, h = id & (Hdim - 1);
    int c = h >> 4, n = h & 15;
    signed char* Wl = W + (size_t)l * (Hdim * Hdim) + (size_t)c * 32768;
    const int*   ep = esrc + (size_t)id * Kfan;
    const float* wp = ew   + (size_t)id * Kfan;
    #pragma unroll 4
    for (int k = 0; k < Kfan; ++k) {
        int s  = ep[k];
        int q  = s >> 6, ks = s & 63;
        int ii   = ((ks >> 5) << 2) | (((ks >> 4) & 1) << 1) | ((ks >> 2) & 1);
        int lane = (((ks >> 3) & 1) << 4) | n;
        signed char* p = Wl + q * 1024 + lane * 32 + ii * 4 + (ks & 3);
        *p = (signed char)(*p + (int)wp[k]);
    }
}

__global__ void wann_encode(const float* __restrict__ x,
                            const float* __restrict__ u_t,
                            ubyte* __restrict__ A) {
    int id = blockIdx.x * 256 + threadIdx.x;
    int b = id >> 11, n = id & (Hdim - 1);
    ubyte s = (u_t[id] < x[id]) ? (ubyte)1 : (ubyte)0;
    int r = b >> 4, m = b & 15;
    int q = n >> 6, ks = n & 63;
    int ii   = ((ks >> 5) << 2) | (((ks >> 4) & 1) << 1) | ((ks >> 2) & 1);
    int lane = (((ks >> 3) & 1) << 4) | m;
    A[(size_t)r * 32768 + q * 1024 + lane * 32 + ii * 4 + (ks & 3)] = s;
}

__global__ __launch_bounds__(256)
void wann_gemm_lif(const signed char* __restrict__ Wl,
                   const ubyte* __restrict__ Ain,
                   ubyte* __restrict__ Aout,
                   ubyte* __restrict__ Splain,
                   float* __restrict__ vh) {
    __shared__ alignas(32) ubyte lds[32768];
    const int tid = threadIdx.x;
    const int r   = blockIdx.y;

    {
        const uint4* src = (const uint4*)(Ain + (size_t)r * 32768);
        uint4* dst = (uint4*)lds;
        #pragma unroll
        for (int i = 0; i < 8; ++i) dst[tid + i * 256] = src[tid + i * 256];
    }
    __syncthreads();

    const int wv    = tid >> 5;
    const int lane  = tid & 31;
    const int ctile = blockIdx.x * 8 + wv;
    const signed char* wp = Wl + (size_t)ctile * 32768 + lane * 32;
    const ubyte*       ap = lds + lane * 32;

    v8i acc = {};
    #pragma unroll 4
    for (int q = 0; q < 32; ++q) {
        v8i a = *(const v8i*)(ap + q * 1024);
        v8i b = *(const v8i*)(wp + q * 1024);
        acc = __builtin_amdgcn_wmma_i32_16x16x64_iu8(
                   false, a,  true, b, acc,
                   false,  false);
    }

    const int n    = lane & 15;
    const int half = lane >> 4;
    const int hcol = ctile * 16 + n;
    const int ks = hcol & 63, q2 = hcol >> 6;
    const int ii = ((ks >> 5) << 2) | (((ks >> 4) & 1) << 1) | ((ks >> 2) & 1);
    const int g  = (ks >> 3) & 1;
    ubyte* outbase = Aout + (size_t)r * 32768 + q2 * 1024 + ii * 4 + (ks & 3);

    #pragma unroll
    for (int j = 0; j < 8; ++j) {
        int m = j + half * 8;
        int b = r * 16 + m;
        float vv  = vh[b * Hdim + hcol];
        vv = 0.9f * vv + 0.1f * (float)acc[j];
        bool s = (vv >= 1.0f);
        vh[b * Hdim + hcol]     = s ? 0.0f : vv;
        Splain[b * Hdim + hcol] = (ubyte)s;
        outbase[(g * 16 + m) * 32] = (ubyte)s;
    }
}

__global__ void wann_output(const ubyte* __restrict__ Splain,
                            const int* __restrict__ eso,
                            const float* __restrict__ ewo,
                            float* __restrict__ vo,
                            float* __restrict__ out) {
    int o = blockIdx.x;
    int b = threadIdx.x;
    const int*   ep = eso + o * KOdim;
    const float* wp = ewo + o * KOdim;
    const ubyte* sr = Splain + (size_t)b * Hdim;
    float sum = 0.0f;
    #pragma unroll 8
    for (int k = 0; k < KOdim; ++k)
        sum += wp[k] * (float)sr[ep[k]];
    float v = vo[b * 16 + o];
    v = 0.9f * v + 0.1f * sum;
    bool s = (v >= 1.0f);
    vo[b * 16 + o] = s ? 0.0f : v;
    if (s) out[b * Odim + o] += 1.0f;
}

extern "C" void kernel_launch(void* const* d_in, const int* in_sizes, int n_in,
                              void* d_out, int out_size, void* d_ws, size_t ws_size,
                              hipStream_t stream) {
    const float* x   = (const float*)d_in[0];
    const float* u   = (const float*)d_in[1];
    const float* ew  = (const float*)d_in[2];
    const float* ewo = (const float*)d_in[3];
    const int*   es  = (const int*)d_in[4];
    const int*   eso = (const int*)d_in[5];
    float* out = (float*)d_out;

    char* ws = (char*)d_ws;
    signed char* W  = (signed char*)(ws + OFF_W);
    ubyte* A0 = (ubyte*)(ws + OFF_A0);
    ubyte* A1 = (ubyte*)(ws + OFF_A1);
    ubyte* SP = (ubyte*)(ws + OFF_SP);
    float* VH = (float*)(ws + OFF_VH);
    float* VO = (float*)(ws + OFF_VO);

    const int n16 = (int)(WS_TOTAL / 16);
    wann_zero<<<(n16 + 255) / 256, 256, 0, stream>>>((uint4*)ws, out, n16,
                                                     Bdim * Odim);
    wann_densify<<<(Ldim * Hdim) / 256, 256, 0, stream>>>(es, ew, W);

    for (int t = 0; t < Tdim; ++t) {
        wann_encode<<<(Bdim * Hdim) / 256, 256, 0, stream>>>(
            x, u + (size_t)t * Bdim * Hdim, A0);
        const ubyte* ain = A0;
        ubyte*       aot = A1;
        for (int l = 0; l < Ldim; ++l) {
            wann_gemm_lif<<<dim3(16, 16), 256, 0, stream>>>(
                W + (size_t)l * (Hdim * Hdim), ain, aot, SP,
                VH + (size_t)l * Bdim * Hdim);
            ubyte* tmp = (ubyte*)ain; ain = aot; aot = tmp;
        }
        wann_output<<<Odim, 256, 0, stream>>>(SP, eso, ewo, VO, out);
    }
}
